// ByteLevelDecoder_72954314490187
// MI455X (gfx1250) — matrix operand order repaired and checked by reading, not compiled, not run on hardware
//
#include <hip/hip_runtime.h>
#include <hip/hip_bf16.h>
#include <math.h>

// ---------------- problem constants ----------------
#define Bn   2
#define Sn   256
#define Hn   1024
#define BHn  384
#define NHn  8
#define HDn  48
#define Pn   4
#define SCn  12
#define Ln   4
#define Vn   258
#define Tn   16
#define Nn   512          // B*S independent sequences
#define VPADn 272         // V padded to multiple of 16
#define SCALE_ 0.14433756729740643f  // 1/sqrt(48)

typedef __attribute__((ext_vector_type(16))) __bf16   v16bf;
typedef __attribute__((ext_vector_type(8)))  float    v8f;
typedef __attribute__((ext_vector_type(4)))  unsigned u32x4;

union Frag { v16bf v; u32x4 u[2]; };

__device__ __forceinline__ unsigned short f2bf(float f) {
    unsigned u = __float_as_uint(f);
    u += 0x7FFFu + ((u >> 16) & 1u);          // round-to-nearest-even
    return (unsigned short)(u >> 16);
}
__device__ __forceinline__ float bf2f(unsigned short h) {
    return __uint_as_float(((unsigned)h) << 16);
}

// ---------------- weight transpose + f32->bf16 ----------------
// dst[b][n][k] = (n < N) ? src[b][k][n] : 0   (dst has Npad rows of length K)
__global__ __launch_bounds__(256) void transpose_pad_bf16(
    const float* __restrict__ src, unsigned short* __restrict__ dst,
    int K, int N, int Npad, int batch)
{
    long long idx = (long long)blockIdx.x * 256 + threadIdx.x;
    long long total = (long long)batch * Npad * K;
    if (idx >= total) return;
    int k = (int)(idx % K);
    long long t = idx / K;
    int n = (int)(t % Npad);
    int b = (int)(t / Npad);
    float v = (n < N) ? src[((long long)b * K + k) * N + n] : 0.f;
    dst[idx] = f2bf(v);
}

// ---------------- elementwise f32 -> bf16 ----------------
__global__ __launch_bounds__(256) void cvt_bf16(
    const float* __restrict__ src, unsigned short* __restrict__ dst, long long n)
{
    long long i = (long long)blockIdx.x * 256 + threadIdx.x;
    if (i < n) dst[i] = f2bf(src[i]);
}

// ---------------- WMMA GEMM: C = act(A @ Bt^T) (+ Res) ----------------
// A : (M x K) bf16 row-major (lda), M multiple of 128.
// Bt: (Npad x K) bf16 row-major (row = output column of C).
// Block = 8 waves; wave w handles M-tile (blockIdx.x*8 + w), all NT column
// tiles starting at n0base + blockIdx.y*NT*16.  B is staged through LDS
// (double-buffered, one barrier per k-step) and shared by all 8 waves.
// Output element index: (m/outLq)*s1 + (m%outLq)*s2 + col.
template <int NT>
__global__ __launch_bounds__(256) void gemm_bf16_wmma(
    const unsigned short* __restrict__ A, int lda,
    const unsigned short* __restrict__ Bt, int ldb,
    int Nreal, int K, int n0base,
    float* __restrict__ Cf, unsigned short* __restrict__ Cb,
    int outLq, long long s1, long long s2,
    const float* __restrict__ Res, long long rs1, long long rs2,
    int actGelu)
{
    __shared__ unsigned short Bsh[2][NT * 16 * 32];   // NT tiles of 16 cols x 32 k

    const int tid   = threadIdx.x;
    const int wave  = tid >> 5;
    const int lane  = tid & 31;
    const int lhalf = lane & 15;
    const int hi    = lane >> 4;

    const int m0 = (blockIdx.x * 8 + wave) << 4;
    const int n0 = n0base + blockIdx.y * (NT * 16);

    const unsigned short* Arow = A + (long long)(m0 + lhalf) * lda;

    // staging role: thread t -> row t/4 (of NT*16), 16-byte segment t%4
    const int srow = tid >> 2;
    const int sseg = tid & 3;
    const bool stager = (NT * 64 >= 256) || (tid < NT * 64);  // folds for NT=4
    const unsigned short* Bsrc =
        Bt + (long long)(n0 + srow) * ldb + sseg * 8;   // + kk at use

    v8f acc[NT];
    #pragma unroll
    for (int nt = 0; nt < NT; ++nt)
        #pragma unroll
        for (int r = 0; r < 8; ++r) acc[nt][r] = 0.f;

    // prime buffer 0
    if (stager) {
        *reinterpret_cast<u32x4*>(&Bsh[0][srow * 32 + sseg * 8]) =
            *reinterpret_cast<const u32x4*>(Bsrc);
    }

    int cur = 0;
    for (int kk = 0; kk < K; kk += 32) {
        __syncthreads();
        // stage next k-slice into the other buffer while computing this one
        if (kk + 32 < K && stager) {
            *reinterpret_cast<u32x4*>(&Bsh[cur ^ 1][srow * 32 + sseg * 8]) =
                *reinterpret_cast<const u32x4*>(Bsrc + kk + 32);
            // warm L2/L0 for the slice after that (global_prefetch_b8)
            if (kk + 64 < K) __builtin_prefetch(Bsrc + kk + 64, 0, 3);
        }

        // A fragment (documented gfx1250 16x32 bf16 layout)
        Frag a;
        const int kA = kk + hi * 8;
        a.u[0] = *reinterpret_cast<const u32x4*>(Arow + kA);
        a.u[1] = *reinterpret_cast<const u32x4*>(Arow + kA + 16);
        if (kk + 32 < K) __builtin_prefetch(Arow + kA + 32, 0, 3);

        // load all NT B fragments first (lets the scheduler batch LDS loads),
        // then issue the NT WMMAs back-to-back
        Frag b[NT];
        #pragma unroll
        for (int nt = 0; nt < NT; ++nt) {
            const unsigned short* brow = &Bsh[cur][(nt * 16 + lhalf) * 32 + hi * 8];
            b[nt].u[0] = *reinterpret_cast<const u32x4*>(brow);
            b[nt].u[1] = *reinterpret_cast<const u32x4*>(brow + 16);
        }
        #pragma unroll
        for (int nt = 0; nt < NT; ++nt) {
            acc[nt] = __builtin_amdgcn_wmma_f32_16x16x32_bf16(
                false, a.v, false, b[nt].v, (short)0, acc[nt], false, false);
        }
        cur ^= 1;
    }

    // store (C/D layout: VGPR r -> M = r + 8*hi, N = lane&15)
    #pragma unroll
    for (int nt = 0; nt < NT; ++nt) {
        int col = n0 + nt * 16 + lhalf;
        if (col >= Nreal) continue;
        #pragma unroll
        for (int r = 0; r < 8; ++r) {
            int m = m0 + hi * 8 + r;
            int mq = m / outLq, mr = m % outLq;
            float v = acc[nt][r];
            if (actGelu) v = 0.5f * v * (1.f + erff(v * 0.7071067811865476f));
            if (Res) v += Res[(long long)mq * rs1 + (long long)mr * rs2 + col];
            long long idx = (long long)mq * s1 + (long long)mr * s2 + col;
            if (Cb) Cb[idx] = f2bf(v);
            else    Cf[idx] = v;
        }
    }
}

// ---------------- RMSNorm -> bf16 (one wave per row of 384) ----------------
__global__ __launch_bounds__(256) void rms_kernel(
    const float* __restrict__ x, const float* __restrict__ gamma,
    unsigned short* __restrict__ out, int Mrows)
{
    int row  = blockIdx.x * 8 + ((int)threadIdx.x >> 5);
    int lane = threadIdx.x & 31;
    if (row >= Mrows) return;
    const float* xr = x + (long long)row * BHn;
    float v[12]; float ss = 0.f;
    #pragma unroll
    for (int j = 0; j < 12; ++j) { v[j] = xr[lane + 32 * j]; ss += v[j] * v[j]; }
    #pragma unroll
    for (int off = 16; off; off >>= 1) ss += __shfl_xor(ss, off, 32);
    float r = rsqrtf(ss * (1.f / BHn) + 1e-5f);
    #pragma unroll
    for (int j = 0; j < 12; ++j) {
        int k = lane + 32 * j;
        out[(long long)row * BHn + k] = f2bf(v[j] * r * gamma[k]);
    }
}

// ---------------- attention over <=16 cached keys ----------------
__global__ __launch_bounds__(256) void attn_kernel(
    const float* __restrict__ q,
    const unsigned short* __restrict__ kc, const unsigned short* __restrict__ vc,
    unsigned short* __restrict__ o_bf, int Mrows, int Lq, int pos, int causal)
{
    int mh = blockIdx.x * 256 + threadIdx.x;
    if (mh >= Mrows * NHn) return;
    int m = mh >> 3, h = mh & 7;
    int seq = m / Lq, p = m % Lq;
    int nk = pos + Lq;
    int limit = causal ? (pos + p) : (nk - 1);

    float qr[HDn];
    const float* qv = q + (long long)m * BHn + h * HDn;
    #pragma unroll
    for (int d = 0; d < HDn; ++d) qr[d] = qv[d];

    float sc[Tn]; float mx = -INFINITY;
    for (int t = 0; t <= limit; ++t) {
        const unsigned short* kr = kc + ((long long)seq * Tn + t) * BHn + h * HDn;
        float s = 0.f;
        #pragma unroll
        for (int d = 0; d < HDn; ++d) s += qr[d] * bf2f(kr[d]);
        s *= SCALE_; sc[t] = s; mx = fmaxf(mx, s);
    }
    float sum = 0.f;
    for (int t = 0; t <= limit; ++t) { sc[t] = __expf(sc[t] - mx); sum += sc[t]; }
    float inv = 1.f / sum;

    float o[HDn];
    #pragma unroll
    for (int d = 0; d < HDn; ++d) o[d] = 0.f;
    for (int t = 0; t <= limit; ++t) {
        float pw = sc[t] * inv;
        const unsigned short* vr = vc + ((long long)seq * Tn + t) * BHn + h * HDn;
        #pragma unroll
        for (int d = 0; d < HDn; ++d) o[d] += pw * bf2f(vr[d]);
    }
    unsigned short* ov = o_bf + (long long)m * BHn + h * HDn;
    #pragma unroll
    for (int d = 0; d < HDn; ++d) ov[d] = f2bf(o[d]);
}

// ---------------- gather xi rows from x_out ----------------
__global__ __launch_bounds__(256) void gather_xi(
    const float* __restrict__ x_out, float* __restrict__ xi,
    int M, int Lq, int startPos)
{
    long long i = (long long)blockIdx.x * 256 + threadIdx.x;
    if (i >= (long long)M * BHn) return;
    int m = (int)(i / BHn), k = (int)(i % BHn);
    xi[i] = x_out[((long long)(m / Lq) * Tn + startPos + (m % Lq)) * BHn + k];
}

// ---------------- gen row: argmax/EOS, masked write-back ----------------
__global__ __launch_bounds__(256) void gen_kernel(
    const float* __restrict__ xi, float* __restrict__ x_out,
    unsigned short* __restrict__ xgen_bf, int* __restrict__ finished,
    int Lq, int step)
{
    int seq  = blockIdx.x * 8 + ((int)threadIdx.x >> 5);
    int lane = threadIdx.x & 31;
    if (seq >= Nn) return;
    const float* g = xi + ((long long)seq * Lq + (Lq - 1)) * BHn;
    float v[12]; float bm = -INFINITY; int bi = 0;
    #pragma unroll
    for (int j = 0; j < 12; ++j) {
        int k = lane + 32 * j;
        float f = g[k]; v[j] = f;
        if (f > bm) { bm = f; bi = k; }
    }
    #pragma unroll
    for (int off = 16; off; off >>= 1) {
        float om = __shfl_xor(bm, off, 32);
        int   oi = __shfl_xor(bi, off, 32);
        if (om > bm || (om == bm && oi < bi)) { bm = om; bi = oi; }
    }
    int unf = finished[seq] ? 0 : 1;
    #pragma unroll
    for (int j = 0; j < 12; ++j) {
        int k = lane + 32 * j;
        float val = unf ? v[j] : 0.f;
        x_out[((long long)seq * Tn + Pn + step) * BHn + k] = val;
        xgen_bf[((long long)seq * SCn + step) * BHn + k] = f2bf(val);
    }
    if (lane == 0 && unf && bi == 257) finished[seq] = 1;
}

// ---------------- host orchestration ----------------
// M multiple of 128; Npad multiple of 16.
static inline void launch_gemm(hipStream_t s,
    const unsigned short* A, int lda, const unsigned short* Bt, int ldb,
    int M, int Npad, int Nreal, int K,
    float* Cf, unsigned short* Cb, int outLq, long long s1, long long s2,
    const float* Res, long long rs1, long long rs2, int gelu)
{
    int gx = M / 128;
    int quads4 = Npad / 64;                    // full 64-column quads
    int remTiles = (Npad - quads4 * 64) / 16;  // leftover 16-column tiles
    if (quads4 > 0) {
        gemm_bf16_wmma<4><<<dim3(gx, quads4), 256, 0, s>>>(
            A, lda, Bt, ldb, Nreal, K, 0,
            Cf, Cb, outLq, s1, s2, Res, rs1, rs2, gelu);
    }
    if (remTiles > 0) {
        gemm_bf16_wmma<1><<<dim3(gx, remTiles), 256, 0, s>>>(
            A, lda, Bt, ldb, Nreal, K, quads4 * 64,
            Cf, Cb, outLq, s1, s2, Res, rs1, rs2, gelu);
    }
}

extern "C" void kernel_launch(void* const* d_in, const int* in_sizes, int n_in,
                              void* d_out, int out_size, void* d_ws, size_t ws_size,
                              hipStream_t stream)
{
    (void)in_sizes; (void)n_in; (void)out_size; (void)ws_size;
    const float* x     = (const float*)d_in[0];
    const float* Wproj = (const float*)d_in[2];
    const float* anorm = (const float*)d_in[3];
    const float* Wq    = (const float*)d_in[4];
    const float* Wk    = (const float*)d_in[5];
    const float* Wv    = (const float*)d_in[6];
    const float* Wo    = (const float*)d_in[7];
    const float* fnorm = (const float*)d_in[8];
    const float* W1    = (const float*)d_in[9];
    const float* W2    = (const float*)d_in[10];
    const float* Wlm   = (const float*)d_in[11];
    float* out = (float*)d_out;

    // ---- workspace carve-up ----
    char* p = (char*)d_ws;
    auto alloc = [&](size_t bytes) -> char* {
        char* r = p; p += (bytes + 255) & ~(size_t)255; return r;
    };
    const long long WW = (long long)BHn * BHn;           // 384*384
    const long long CACHE = (long long)Nn * Tn * BHn;    // per-layer KV elems

    unsigned short* WprojT = (unsigned short*)alloc(2ull * 1536 * 1024);
    unsigned short* WqT    = (unsigned short*)alloc(2ull * Ln * WW);
    unsigned short* WkT    = (unsigned short*)alloc(2ull * Ln * WW);
    unsigned short* WvT    = (unsigned short*)alloc(2ull * Ln * WW);
    unsigned short* WoT    = (unsigned short*)alloc(2ull * Ln * WW);
    unsigned short* W1T    = (unsigned short*)alloc(2ull * Ln * 1536 * 384);
    unsigned short* W2T    = (unsigned short*)alloc(2ull * Ln * 384 * 1536);
    unsigned short* WlmT   = (unsigned short*)alloc(2ull * VPADn * BHn);
    unsigned short* x_bf   = (unsigned short*)alloc(2ull * Nn * Hn);
    unsigned short* h_bf   = (unsigned short*)alloc(2ull * 2048 * BHn);
    unsigned short* o_bf   = (unsigned short*)alloc(2ull * 2048 * BHn);
    unsigned short* g_bf   = (unsigned short*)alloc(2ull * 2048 * 1536);
    unsigned short* kc     = (unsigned short*)alloc(2ull * Ln * CACHE);
    unsigned short* vc     = (unsigned short*)alloc(2ull * Ln * CACHE);
    unsigned short* xgen   = (unsigned short*)alloc(2ull * Nn * SCn * BHn);
    float* x_out = (float*)alloc(4ull * Nn * Tn * BHn);
    float* xi    = (float*)alloc(4ull * 2048 * BHn);
    float* qbuf  = (float*)alloc(4ull * 2048 * BHn);
    int*   fin   = (int*)alloc(4ull * Nn);

    auto tlaunch = [&](const float* src, unsigned short* dst, int K, int Ncols,
                       int Npad, int batch) {
        long long tot = (long long)batch * Npad * K;
        transpose_pad_bf16<<<(int)((tot + 255) / 256), 256, 0, stream>>>(
            src, dst, K, Ncols, Npad, batch);
    };

    // ---- weight prep (f32 -> bf16, transposed so B-operand rows are contiguous)
    tlaunch(Wproj, WprojT, Hn, 1536, 1536, 1);
    tlaunch(Wq, WqT, BHn, BHn, BHn, Ln);
    tlaunch(Wk, WkT, BHn, BHn, BHn, Ln);
    tlaunch(Wv, WvT, BHn, BHn, BHn, Ln);
    tlaunch(Wo, WoT, BHn, BHn, BHn, Ln);
    tlaunch(W1, W1T, BHn, 1536, 1536, Ln);
    tlaunch(W2, W2T, 1536, BHn, BHn, Ln);
    tlaunch(Wlm, WlmT, BHn, Vn, VPADn, 1);

    // input -> bf16; finished flags -> 0
    {
        long long nx = (long long)Nn * Hn;
        cvt_bf16<<<(int)((nx + 255) / 256), 256, 0, stream>>>(x, x_bf, nx);
    }
    hipMemsetAsync(fin, 0, sizeof(int) * Nn, stream);

    // ---- input projection: (512 x 1024) @ (1024 x 1536) -> x_out[:, 0:4, :]
    // x_out row n is contiguous 16*384 floats; first 1536 are positions 0..3.
    launch_gemm(stream, x_bf, Hn, WprojT, Hn, Nn, 1536, 1536, Hn,
                x_out, nullptr, 1, (long long)Tn * BHn, 0, nullptr, 0, 0, 0);

    // ---- decode loop ----
    for (int i = 0; i < SCn; ++i) {
        int Lq  = (i == 0) ? Pn : 1;
        int pos = (i == 0) ? 0 : (Pn + i - 1);
        int M   = Nn * Lq;

        gather_xi<<<(int)(((long long)M * BHn + 255) / 256), 256, 0, stream>>>(
            x_out, xi, M, Lq, pos);

        for (int l = 0; l < Ln; ++l) {
            const unsigned short* wq = WqT + (long long)l * WW;
            const unsigned short* wk = WkT + (long long)l * WW;
            const unsigned short* wv = WvT + (long long)l * WW;
            const unsigned short* wo = WoT + (long long)l * WW;
            const unsigned short* w1 = W1T + (long long)l * 1536 * 384;
            const unsigned short* w2 = W2T + (long long)l * 384 * 1536;
            unsigned short* kcl = kc + (long long)l * CACHE;
            unsigned short* vcl = vc + (long long)l * CACHE;

            // h = rms(xi, attn_norm[l])  -> bf16
            rms_kernel<<<(M + 7) / 8, 256, 0, stream>>>(xi, anorm + l * BHn, h_bf, M);

            // q (f32), k/v (bf16 scattered into cache at position pos)
            launch_gemm(stream, h_bf, BHn, wq, BHn, M, BHn, BHn, BHn,
                        qbuf, nullptr, 1, BHn, 0, nullptr, 0, 0, 0);
            launch_gemm(stream, h_bf, BHn, wk, BHn, M, BHn, BHn, BHn,
                        nullptr, kcl + (long long)pos * BHn, Lq,
                        (long long)Tn * BHn, BHn, nullptr, 0, 0, 0);
            launch_gemm(stream, h_bf, BHn, wv, BHn, M, BHn, BHn, BHn,
                        nullptr, vcl + (long long)pos * BHn, Lq,
                        (long long)Tn * BHn, BHn, nullptr, 0, 0, 0);

            // attention over cached keys
            attn_kernel<<<(M * NHn + 255) / 256, 256, 0, stream>>>(
                qbuf, kcl, vcl, o_bf, M, Lq, pos, (i == 0) ? 1 : 0);

            // xi += o @ Wo
            launch_gemm(stream, o_bf, BHn, wo, BHn, M, BHn, BHn, BHn,
                        xi, nullptr, 1, BHn, 0, xi, BHn, 0, 0);

            // FFN: g = gelu(rms(xi) @ W1) ; xi += g @ W2
            rms_kernel<<<(M + 7) / 8, 256, 0, stream>>>(xi, fnorm + l * BHn, h_bf, M);
            launch_gemm(stream, h_bf, BHn, w1, BHn, M, 1536, 1536, BHn,
                        nullptr, g_bf, 1, 1536, 0, nullptr, 0, 0, 1);
            launch_gemm(stream, g_bf, 1536, w2, 1536, M, BHn, BHn, 1536,
                        xi, nullptr, 1, BHn, 0, xi, BHn, 0, 0);
        }

        gen_kernel<<<Nn / 8, 256, 0, stream>>>(xi, x_out, xgen, fin, Lq, i);
    }

    // ---- LM head: (512*12 x 384) @ (384 x 258) -> out (B,S,S_C,V)
    launch_gemm(stream, xgen, BHn, WlmT, BHn, Nn * SCn, VPADn, Vn, BHn,
                out, nullptr, 1, Vn, 0, nullptr, 0, 0, 0);
}
